// MambaBlock_10737418240378
// MI455X (gfx1250) — matrix operand order repaired and checked by reading, not compiled, not run on hardware
//
#include <hip/hip_runtime.h>
#include <hip/hip_bf16.h>

// ---------------- problem constants ----------------
#define BATCH     8
#define DMODEL    256
#define DSSM      512
#define DSTATE    64
#define NHEADS    8
#define HEADDIM   64
#define CONVDIM   640            // DSSM + 2*DSTATE
#define DPROJ     1160           // 2*D_INNER + 2*D_STATE + NHEADS
#define DPROJ_PAD 1280           // padded to multiple of 128 -> guard-free GEMM
#define SEQL      4096           // 64*64 flattened spatial
#define PADHW     66             // 64 + 2 halo
#define PADSP     (PADHW*PADHW)  // 4356
#define EPSF      1e-5f

// ---------------- WMMA bf16 plumbing ----------------
typedef __attribute__((ext_vector_type(16))) __bf16 v16bf;
typedef __attribute__((ext_vector_type(8)))  float  v8f;

__device__ __forceinline__ float silu_f(float x) { return x / (1.0f + __expf(-x)); }

// A fragment (16x32 bf16) from row-major memory, row stride ld (elements).
// ISA: lanes 0-15 -> M=lane, K=0..7 (v0..3) & K=16..23 (v4..7);
//      lanes 16-31 -> same M, K=8..15 & K=24..31.
__device__ __forceinline__ v16bf frag_a_ld(const __bf16* p, int ld) {
  const int lane = threadIdx.x & 31;
  const int m  = lane & 15;
  const int kb = (lane >> 4) << 3;
  const __bf16* r = p + (size_t)m * ld + kb;
  v16bf f;
#pragma unroll
  for (int i = 0; i < 8; ++i) f[i] = r[i];
#pragma unroll
  for (int i = 0; i < 8; ++i) f[i + 8] = r[16 + i];
  return f;
}

// B fragment (32x16 bf16) from memory where each output column n has its 32
// K-values contiguous (column stride ld elements).
// ISA: lanes 0-15 -> N=lane, K=0..7 & K=16..23; lanes 16-31 -> N=lane-16, K=8..15 & K=24..31.
__device__ __forceinline__ v16bf frag_b_ld(const __bf16* p, int ld) {
  const int lane = threadIdx.x & 31;
  const int n  = lane & 15;
  const int kb = (lane >> 4) << 3;
  const __bf16* r = p + (size_t)n * ld + kb;
  v16bf f;
#pragma unroll
  for (int i = 0; i < 8; ++i) f[i] = r[i];
#pragma unroll
  for (int i = 0; i < 8; ++i) f[i + 8] = r[16 + i];
  return f;
}

__device__ __forceinline__ v8f wmma_bf16(v16bf a, v16bf b, v8f c) {
  return __builtin_amdgcn_wmma_f32_16x16x32_bf16(false, a, false, b, (short)0, c,
                                                 false, false);
}

// ---------------- gfx1250 async global->LDS copy (ASYNCcnt path) ----------------
__device__ __forceinline__ void async_copy_b128(unsigned lds_off, const void* gptr) {
  asm volatile("global_load_async_to_lds_b128 %0, %1, off"
               :: "v"(lds_off), "v"((unsigned long long)(size_t)gptr)
               : "memory");
}
__device__ __forceinline__ void wait_async0() {   // all async copies landed
  asm volatile("s_wait_asynccnt 0x0" ::: "memory");
}
__device__ __forceinline__ void wait_async1() {   // all but newest landed (double buffer)
  asm volatile("s_wait_asynccnt 0x1" ::: "memory");
}
__device__ __forceinline__ unsigned lds_addr_of(const void* p) {
  return (unsigned)(size_t)p;   // generic LDS pointer: low 32 bits = LDS offset
}

// ---------------- pack kernels ----------------
// u (b,c,l) fp32 -> uT (b,l,c) bf16 : makes in-proj B fragments contiguous.
__global__ void k_pack_ut(const float* __restrict__ u, __bf16* __restrict__ uT) {
  int j = blockIdx.x * 256 + threadIdx.x;            // over BATCH*SEQL*DMODEL
  if (j >= BATCH * SEQL * DMODEL) return;
  int b = j >> 20, l = (j >> 8) & (SEQL - 1), c = j & (DMODEL - 1);
  uT[j] = (__bf16)u[((size_t)(b * DMODEL + c)) * SEQL + l];
}

__global__ void k_pack_win(const float* __restrict__ W, __bf16* __restrict__ ob) {
  int i = blockIdx.x * 256 + threadIdx.x;            // over DPROJ_PAD*DMODEL
  if (i >= DPROJ_PAD * DMODEL) return;
  int row = i / DMODEL;
  ob[i] = (__bf16)(row < DPROJ ? W[i] : 0.0f);
}

// W_out (256,512,3,3) -> Wb[p][co][ci]
__global__ void k_pack_wout(const float* __restrict__ W, __bf16* __restrict__ ob) {
  int i = blockIdx.x * 256 + threadIdx.x;
  if (i >= 9 * 256 * DSSM) return;
  int p  = i / (256 * DSSM);
  int co = (i / DSSM) % 256;
  int ci = i % DSSM;
  ob[i] = (__bf16)W[(size_t)(co * DSSM + ci) * 9 + p];
}

// ---------------- in-projection GEMM: zx[b][m][l] = W_in[m][k]*u[b][k][l] ----------------
// grid (SEQL/32, DPROJ_PAD/128, BATCH), 256 thr = 8 waves; per wave 16 co x 32 l.
// B tile (32k x 32l) double-buffered via async b128 into LDS [l][k].
__global__ __launch_bounds__(256) void k_inproj(const __bf16* __restrict__ Win,
                                                const __bf16* __restrict__ uT,
                                                float* __restrict__ zx) {
  __shared__ __align__(16) __bf16 bt[2][32 * 32];    // ping-pong [l 0..31][k 0..31]
  const int b  = blockIdx.z;
  const int l0 = blockIdx.x * 32;
  const int t  = threadIdx.x;
  const int m0 = blockIdx.y * 128 + (t >> 5) * 16;
  const unsigned bt0 = lds_addr_of(&bt[0][0]);
  const __bf16* ubase = uT + ((size_t)b * SEQL + l0) * DMODEL;

  auto issue = [&](int s) {                          // stage s -> k0 = s*32
    if (t < 128) {                                   // 32 rows x 64B, 4 lanes/row
      int li = t >> 2, q = t & 3;
      async_copy_b128(bt0 + (unsigned)((s & 1) * 2048 + li * 64 + q * 16),
                      ubase + (size_t)li * DMODEL + s * 32 + q * 8);
    }
  };

  v8f acc0 = {}, acc1 = {};
  issue(0);
  for (int s = 0; s < DMODEL / 32; ++s) {
    if (s + 1 < DMODEL / 32) issue(s + 1);           // stream next stage behind compute
    wait_async1();                                   // oldest copy landed
    __syncthreads();
    const __bf16* bs = &bt[s & 1][0];
    v16bf a   = frag_a_ld(Win + (size_t)m0 * DMODEL + s * 32, DMODEL);
    v16bf bb0 = frag_b_ld(bs, 32);
    v16bf bb1 = frag_b_ld(bs + 16 * 32, 32);
    acc0 = wmma_bf16(a, bb0, acc0);
    acc1 = wmma_bf16(a, bb1, acc1);
    __syncthreads();                                 // reads done before buffer reuse
  }
  const int lane = t & 31, col = lane & 15, rb = (lane >> 4) * 8;
  float* o = zx + ((size_t)b * DPROJ_PAD + m0) * SEQL + l0;
#pragma unroll
  for (int r = 0; r < 8; ++r) {
    o[(size_t)(rb + r) * SEQL + col]      = acc0[r];
    o[(size_t)(rb + r) * SEQL + col + 16] = acc1[r];
  }
}

// ---------------- depthwise causal conv1d kernels ----------------
__global__ void k_conv_dt(const float* __restrict__ zx, const float* __restrict__ w,
                          const float* __restrict__ bias, const float* __restrict__ A_log,
                          float* __restrict__ dA) {
  int l = blockIdx.x * 256 + threadIdx.x;
  int h = blockIdx.y, b = blockIdx.z;
  const float* in = zx + ((size_t)b * DPROJ_PAD + (DSSM + CONVDIM) + h) * SEQL;
  float s = bias[h];
#pragma unroll
  for (int j = 0; j < 4; ++j) { int li = l - 3 + j; if (li >= 0) s += w[h * 4 + j] * in[li]; }
  dA[((size_t)b * NHEADS + h) * SEQL + l] = silu_f(s) * (-__expf(A_log[h]));
}

// xBC conv: emits V (xv), V*dA (xs) in [c][l]; B in [n][l]; C transposed to [l][n].
__global__ void k_conv_xbc(const float* __restrict__ zx, const float* __restrict__ w,
                           const float* __restrict__ bias, const float* __restrict__ dA,
                           __bf16* __restrict__ xs, __bf16* __restrict__ xv,
                           __bf16* __restrict__ Bm, __bf16* __restrict__ CmT) {
  int l = blockIdx.x * 256 + threadIdx.x;
  int c = blockIdx.y, b = blockIdx.z;
  const float* in = zx + ((size_t)b * DPROJ_PAD + DSSM + c) * SEQL;
  float s = bias[c];
#pragma unroll
  for (int j = 0; j < 3; ++j) { int li = l - 2 + j; if (li >= 0) s += w[c * 3 + j] * in[li]; }
  float v = silu_f(s);
  if (c < DSSM) {
    xv[((size_t)b * DSSM + c) * SEQL + l] = (__bf16)v;
    xs[((size_t)b * DSSM + c) * SEQL + l] =
        (__bf16)(v * dA[((size_t)b * NHEADS + (c >> 6)) * SEQL + l]);
  } else if (c < DSSM + DSTATE) {
    Bm[((size_t)b * DSTATE + (c - DSSM)) * SEQL + l] = (__bf16)v;
  } else {
    CmT[((size_t)b * SEQL + l) * DSTATE + (c - DSSM - DSTATE)] = (__bf16)v;
  }
}

__global__ void k_conv_z(const float* __restrict__ zx, const float* __restrict__ w,
                         const float* __restrict__ bias, __bf16* __restrict__ zc) {
  int l = blockIdx.x * 256 + threadIdx.x;
  int c = blockIdx.y, b = blockIdx.z;
  const float* in = zx + ((size_t)b * DPROJ_PAD + c) * SEQL;
  float s = bias[c];
#pragma unroll
  for (int j = 0; j < 3; ++j) { int li = l - 2 + j; if (li >= 0) s += w[c * 3 + j] * in[li]; }
  zc[((size_t)b * DSSM + c) * SEQL + l] = (__bf16)silu_f(s);
}

// ---------------- KV GEMM: KV[b,h][n][d] = sum_l B[n,l]*Vs[d,l] (LDS-free) ----------------
__global__ __launch_bounds__(256) void k_kv(const __bf16* __restrict__ Bm,
                                            const __bf16* __restrict__ xs,
                                            float* __restrict__ KV) {
  const int bh = blockIdx.x, b = bh >> 3, h = bh & 7;
  const int wv = threadIdx.x >> 5;
  const int n0 = (wv & 3) * 16;
  const int d0 = (wv >> 2) * 16;
  const __bf16* Ab = Bm + ((size_t)b * DSTATE + n0) * SEQL;
  const __bf16* B0 = xs + ((size_t)(b * DSSM + h * HEADDIM + d0)) * SEQL;
  const __bf16* B1 = B0 + (size_t)32 * SEQL;
  v8f acc0 = {}, acc1 = {};
  for (int l0 = 0; l0 < SEQL; l0 += 32) {
    __builtin_prefetch(Ab + l0 + 256, 0, 1);
    v16bf a  = frag_a_ld(Ab + l0, SEQL);
    v16bf f0 = frag_b_ld(B0 + l0, SEQL);
    v16bf f1 = frag_b_ld(B1 + l0, SEQL);
    acc0 = wmma_bf16(a, f0, acc0);
    acc1 = wmma_bf16(a, f1, acc1);
  }
  const int lane = threadIdx.x & 31, col = lane & 15, rb = (lane >> 4) * 8;
  float* o = KV + (size_t)bh * DSTATE * HEADDIM;
#pragma unroll
  for (int r = 0; r < 8; ++r) {
    int n = n0 + rb + r;
    o[n * HEADDIM + d0 + col]      = acc0[r];
    o[n * HEADDIM + d0 + 32 + col] = acc1[r];
  }
}

// KV (n,d) fp32 -> KVT (d,n) bf16
__global__ void k_kvt(const float* __restrict__ KV, __bf16* __restrict__ KVT) {
  int bh = blockIdx.x, t = threadIdx.x;
  const float* src = KV + (size_t)bh * 4096;
  __bf16* dst = KVT + (size_t)bh * 4096;
  for (int e = t; e < 4096; e += 256) {
    int n = e >> 6, d = e & 63;
    dst[d * 64 + n] = (__bf16)src[e];
  }
}

// ---------------- Y GEMM: yg[b][l][h*64+d] = sum_n C[n,l]*KV[n,d] (LDS-free) ----------------
__global__ __launch_bounds__(256) void k_y(const __bf16* __restrict__ CmT,
                                           const __bf16* __restrict__ KVT,
                                           float* __restrict__ yg) {
  const int bh = blockIdx.y, b = bh >> 3, h = bh & 7;
  const int t = threadIdx.x, wv = t >> 5;
  const int l0w = blockIdx.x * 128 + wv * 16;
  const __bf16* Ab = CmT + ((size_t)b * SEQL + l0w) * DSTATE;
  const __bf16* Bb = KVT + (size_t)bh * 4096;
  v8f acc[4] = {};
#pragma unroll
  for (int k0 = 0; k0 < DSTATE; k0 += 32) {
    v16bf a = frag_a_ld(Ab + k0, DSTATE);
#pragma unroll
    for (int dtile = 0; dtile < 4; ++dtile) {
      v16bf bb = frag_b_ld(Bb + dtile * 16 * 64 + k0, 64);
      acc[dtile] = wmma_bf16(a, bb, acc[dtile]);
    }
  }
  const int lane = t & 31, col = lane & 15, rb = (lane >> 4) * 8;
#pragma unroll
  for (int dtile = 0; dtile < 4; ++dtile)
#pragma unroll
    for (int r = 0; r < 8; ++r) {
      int l = l0w + rb + r;
      yg[((size_t)b * SEQL + l) * DSSM + h * HEADDIM + dtile * 16 + col] = acc[dtile][r];
    }
}

// ---------------- zero the padded channel-last image ----------------
__global__ void k_zero_yp(__bf16* __restrict__ yp, int n) {
  int i = blockIdx.x * 256 + threadIdx.x;
  if (i < n) yp[i] = (__bf16)0.0f;
}

// ---------------- gate + D*V skip + RMS norm -> yp[b][(h+1)*66+(w+1)][c] ----------------
__global__ __launch_bounds__(128) void k_gatenorm(const float* __restrict__ yg,
                                                  const __bf16* __restrict__ xv,
                                                  const __bf16* __restrict__ zc,
                                                  const float* __restrict__ Dp,
                                                  const float* __restrict__ norm_w,
                                                  __bf16* __restrict__ yp) {
  const int l = blockIdx.x, b = blockIdx.y, t = threadIdx.x;
  __shared__ float red[128];
  float v[4], ss = 0.0f;
#pragma unroll
  for (int i = 0; i < 4; ++i) {
    int c = t + i * 128;
    float val = yg[((size_t)b * SEQL + l) * DSSM + c] +
                Dp[c >> 6] * (float)xv[((size_t)b * DSSM + c) * SEQL + l];
    val *= silu_f((float)zc[((size_t)b * DSSM + c) * SEQL + l]);
    v[i] = val;
    ss += val * val;
  }
  red[t] = ss;
  __syncthreads();
  for (int o = 64; o > 0; o >>= 1) { if (t < o) red[t] += red[t + o]; __syncthreads(); }
  float rs = rsqrtf(red[0] / (float)DSSM + EPSF);
  const int hh = l >> 6, ww = l & 63;
  __bf16* o = yp + ((size_t)b * PADSP + (size_t)(hh + 1) * PADHW + (ww + 1)) * DSSM;
#pragma unroll
  for (int i = 0; i < 4; ++i) {
    int c = t + i * 128;
    o[c] = (__bf16)(v[i] * rs * norm_w[c]);
  }
}

// ---------------- 3x3 conv2d as implicit GEMM over padded channel-last y ----------------
// grid (SEQL/32, 2, BATCH), 256 thr = 8 waves x 16 co; per wave two 16-l tiles.
// 144-stage pipeline: B tile (32 ci x 32 l) double-buffered via async b128.
__global__ __launch_bounds__(256) void k_conv2d(const __bf16* __restrict__ Wb,
                                                const __bf16* __restrict__ yp,
                                                float* __restrict__ out) {
  __shared__ __align__(16) __bf16 bt[2][32 * 32];    // ping-pong [l 0..31][ci 0..31]
  const int b  = blockIdx.z;
  const int l0 = blockIdx.x * 32;
  const int hh = l0 >> 6, w0 = l0 & 63;
  const int t  = threadIdx.x;
  const int m0 = blockIdx.y * 128 + (t >> 5) * 16;
  const unsigned bt0 = lds_addr_of(&bt[0][0]);
  const __bf16* ybase = yp + (size_t)b * PADSP * DSSM;

  auto issue = [&](int s) {                          // stage s: p = s/16, c0 = (s%16)*32
    if (t < 128) {
      int p = s >> 4, c0 = (s & 15) << 5;
      int dy = p / 3 - 1, dx = p % 3 - 1;
      size_t rowbase = ((size_t)(hh + dy + 1) * PADHW + (w0 + dx + 1)) * DSSM;
      int li = t >> 2, q = t & 3;
      async_copy_b128(bt0 + (unsigned)((s & 1) * 2048 + li * 64 + q * 16),
                      ybase + rowbase + (size_t)li * DSSM + c0 + q * 8);
    }
  };

  v8f acc0 = {}, acc1 = {};
  issue(0);
  for (int s = 0; s < 144; ++s) {
    if (s + 1 < 144) issue(s + 1);                   // stream next tile behind WMMAs
    wait_async1();                                   // stage s landed
    __syncthreads();
    const int p = s >> 4, c0 = (s & 15) << 5;
    __builtin_prefetch(Wb + ((size_t)(p * 256 + m0)) * DSSM + c0 + 32, 0, 1);
    const __bf16* bs = &bt[s & 1][0];
    v16bf a   = frag_a_ld(Wb + ((size_t)(p * 256 + m0)) * DSSM + c0, DSSM);
    v16bf bb0 = frag_b_ld(bs, 32);
    v16bf bb1 = frag_b_ld(bs + 16 * 32, 32);
    acc0 = wmma_bf16(a, bb0, acc0);
    acc1 = wmma_bf16(a, bb1, acc1);
    __syncthreads();                                 // reads done before buffer reuse
  }
  const int lane = t & 31, col = lane & 15, rb = (lane >> 4) * 8;
#pragma unroll
  for (int r = 0; r < 8; ++r) {
    float* o = out + ((size_t)b * 256 + m0 + rb + r) * SEQL + l0;
    o[col]      = acc0[r];
    o[col + 16] = acc1[r];
  }
}

// ---------------- host: workspace carve + launch sequence ----------------
extern "C" void kernel_launch(void* const* d_in, const int* in_sizes, int n_in,
                              void* d_out, int out_size, void* d_ws, size_t ws_size,
                              hipStream_t stream) {
  (void)in_sizes; (void)n_in; (void)out_size; (void)ws_size;
  const float* u       = (const float*)d_in[0];
  const float* W_in    = (const float*)d_in[1];
  const float* w_xBC   = (const float*)d_in[2];
  const float* b_xBC   = (const float*)d_in[3];
  const float* w_z     = (const float*)d_in[4];
  const float* b_z     = (const float*)d_in[5];
  const float* w_dt    = (const float*)d_in[6];
  const float* dt_bias = (const float*)d_in[7];
  const float* A_log   = (const float*)d_in[8];
  const float* Dp      = (const float*)d_in[9];
  const float* norm_w  = (const float*)d_in[10];
  const float* W_out   = (const float*)d_in[11];
  float* out = (float*)d_out;

  char* ws = (char*)d_ws;
  size_t off = 0;
  auto alloc = [&](size_t bytes) -> void* {
    void* p = (void*)(ws + off);
    off = (off + bytes + 255) & ~(size_t)255;
    return p;
  };
  __bf16* uT    = (__bf16*)alloc((size_t)BATCH * SEQL * DMODEL * 2);
  __bf16* WinB  = (__bf16*)alloc((size_t)DPROJ_PAD * DMODEL * 2);
  __bf16* WoutB = (__bf16*)alloc((size_t)9 * 256 * DSSM * 2);
  float*  zx    = (float*) alloc((size_t)BATCH * DPROJ_PAD * SEQL * 4);
  float*  dA    = (float*) alloc((size_t)BATCH * NHEADS * SEQL * 4);
  __bf16* xs    = (__bf16*)alloc((size_t)BATCH * DSSM * SEQL * 2);
  __bf16* xv    = (__bf16*)alloc((size_t)BATCH * DSSM * SEQL * 2);
  __bf16* Bm    = (__bf16*)alloc((size_t)BATCH * DSTATE * SEQL * 2);
  __bf16* CmT   = (__bf16*)alloc((size_t)BATCH * SEQL * DSTATE * 2);
  __bf16* zc    = (__bf16*)alloc((size_t)BATCH * DSSM * SEQL * 2);
  float*  KV    = (float*) alloc((size_t)BATCH * NHEADS * DSTATE * HEADDIM * 4);
  __bf16* KVT   = (__bf16*)alloc((size_t)BATCH * NHEADS * DSTATE * HEADDIM * 2);
  float*  yg    = (float*) alloc((size_t)BATCH * SEQL * DSSM * 4);
  __bf16* yp    = (__bf16*)alloc((size_t)BATCH * PADSP * DSSM * 2);

  const int nu = BATCH * SEQL * DMODEL;
  k_pack_ut  <<<(nu + 255) / 256, 256, 0, stream>>>(u, uT);
  k_pack_win <<<(DPROJ_PAD * DMODEL + 255) / 256, 256, 0, stream>>>(W_in, WinB);
  k_pack_wout<<<(9 * 256 * DSSM + 255) / 256, 256, 0, stream>>>(W_out, WoutB);

  k_inproj<<<dim3(SEQL / 32, DPROJ_PAD / 128, BATCH), 256, 0, stream>>>(WinB, uT, zx);

  k_conv_dt <<<dim3(SEQL / 256, NHEADS, BATCH), 256, 0, stream>>>(zx, w_dt, dt_bias, A_log, dA);
  k_conv_xbc<<<dim3(SEQL / 256, CONVDIM, BATCH), 256, 0, stream>>>(zx, w_xBC, b_xBC, dA, xs, xv, Bm, CmT);
  k_conv_z  <<<dim3(SEQL / 256, DSSM, BATCH), 256, 0, stream>>>(zx, w_z, b_z, zc);

  k_kv <<<BATCH * NHEADS, 256, 0, stream>>>(Bm, xs, KV);
  k_kvt<<<BATCH * NHEADS, 256, 0, stream>>>(KV, KVT);
  k_y  <<<dim3(SEQL / 128, BATCH * NHEADS), 256, 0, stream>>>(CmT, KVT, yg);

  const int np = BATCH * PADSP * DSSM;
  k_zero_yp<<<(np + 255) / 256, 256, 0, stream>>>(yp, np);
  k_gatenorm<<<dim3(SEQL, BATCH), 128, 0, stream>>>(yg, xv, zc, Dp, norm_w, yp);

  k_conv2d<<<dim3(SEQL / 32, 2, BATCH), 256, 0, stream>>>(WoutB, yp, out);
}
